// PyramidNeRF_11587821765172
// MI455X (gfx1250) — compile-verified, matrix operand order checked by reading, not run on hardware
//
#include <hip/hip_runtime.h>
#include <hip/hip_bf16.h>
#include <math.h>

typedef _Float16 v16h __attribute__((ext_vector_type(16)));
typedef _Float16 h8   __attribute__((ext_vector_type(8)));
typedef float    v8f  __attribute__((ext_vector_type(8)));

#define NPTS  262144
#define FEAT  128
#define HID   256
#define SX    264      // LDS row stride (f16) for 256-wide buffers (+8 pad)
#define SF    136      // LDS row stride (f16) for 128-wide feats (+8 pad)
#define MTILE 64       // points per block (4 M-tiles)

// f16-element offsets of swizzled weights inside the weight region of d_ws
static constexpr size_t kWOff[11] = {
    0,        // W0   128x256
    32768,    // W1   256x256
    98304,    // W2
    163840,   // W3
    229376,   // W4   384x256 (skip layer)
    327680,   // W5
    393216,   // W6
    458752,   // W7
    524288,   // Wf   256x256
    589824,   // Wv0a 256x128 (first 256 rows of Wv0)
    622592    // total f16 elements
};

// ---------------------------------------------------------------------------
// Weight swizzle: f32 W (K x Nout, row-major) -> f16 WMMA B-fragments.
//   n = nt*16 + (L&15); half = L>>4
//   k_local(i) = i + (i>=8 ? 8 : 0) + half*8     (mirrors ISA 16-bit A layout)
// Stored linearly as (((kt*NT + nt)*32 + lane)*16 + i).
// ---------------------------------------------------------------------------
__global__ void swizzle_kernel(const float* __restrict__ W,
                               _Float16* __restrict__ dst, int K, int Nout) {
  int tid = blockIdx.x * blockDim.x + threadIdx.x;
  if (tid >= K * Nout) return;
  int i    = tid & 15;
  int lane = (tid >> 4) & 31;
  int lt   = tid >> 9;
  int NT   = Nout >> 4;
  int nt   = lt % NT;
  int kt   = lt / NT;
  int half = lane >> 4;
  int k = kt * 32 + i + ((i >= 8) ? 8 : 0) + half * 8;
  int n = nt * 16 + (lane & 15);
  dst[tid] = (_Float16)W[(size_t)k * Nout + n];
}

// ---------------------------------------------------------------------------
// Trilinear sampling: grid (C=32, D,H,W), align_corners=True + border clamp.
// ---------------------------------------------------------------------------
template <int R>
__device__ __forceinline__ void sample_level(const float* __restrict__ g,
                                             float cx, float cy, float cz,
                                             _Float16* __restrict__ dst) {
  const float s = (float)(R - 1);
  float tx = fminf(fmaxf(cx * s, 0.f), s);
  float ty = fminf(fmaxf(cy * s, 0.f), s);
  float tz = fminf(fmaxf(cz * s, 0.f), s);
  float fx0 = floorf(tx), fy0 = floorf(ty), fz0 = floorf(tz);
  int x0 = (int)fx0, y0 = (int)fy0, z0 = (int)fz0;
  int x1 = (x0 + 1 < R) ? x0 + 1 : R - 1;
  int y1 = (y0 + 1 < R) ? y0 + 1 : R - 1;
  int z1 = (z0 + 1 < R) ? z0 + 1 : R - 1;
  float fx = tx - fx0, fy = ty - fy0, fz = tz - fz0;

  float acc[32];
#pragma unroll
  for (int c = 0; c < 32; ++c) acc[c] = 0.f;

  const size_t R2 = (size_t)R * R;
  const size_t R3 = R2 * (size_t)R;
#pragma unroll
  for (int corner = 0; corner < 8; ++corner) {
    int xi = (corner & 1) ? x1 : x0;
    int yi = (corner & 2) ? y1 : y0;
    int zi = (corner & 4) ? z1 : z0;
    float w = ((corner & 1) ? fx : 1.f - fx) *
              ((corner & 2) ? fy : 1.f - fy) *
              ((corner & 4) ? fz : 1.f - fz);
    const float* base = g + (size_t)zi * R2 + (size_t)yi * (size_t)R + (size_t)xi;
#pragma unroll
    for (int c = 0; c < 32; ++c) acc[c] += w * base[(size_t)c * R3];
  }
#pragma unroll
  for (int c = 0; c < 32; ++c) dst[c] = (_Float16)acc[c];
}

__global__ void sample_kernel(const float* __restrict__ coords,
                              const float* __restrict__ g0, const float* __restrict__ g1,
                              const float* __restrict__ g2, const float* __restrict__ g3,
                              _Float16* __restrict__ feats) {
  int p = blockIdx.x * blockDim.x + threadIdx.x;
  if (p >= NPTS) return;
  float cx = coords[p * 3 + 0], cy = coords[p * 3 + 1], cz = coords[p * 3 + 2];
  _Float16* dst = feats + (size_t)p * FEAT;
  sample_level<16 >(g0, cx, cy, cz, dst + 0);
  sample_level<32 >(g1, cx, cy, cz, dst + 32);
  sample_level<64 >(g2, cx, cy, cz, dst + 64);
  sample_level<128>(g3, cx, cy, cz, dst + 96);
}

// ---------------------------------------------------------------------------
// Fused WMMA MLP: block = 64 points, 8 wave32 waves, 4x2 tiles per wave.
// ---------------------------------------------------------------------------
struct MlpArgs {
  const _Float16* feats;
  const _Float16* w;       // swizzled weight base
  const float* vd;         // view dirs (N x 3)
  const float* b[8];
  const float* Wd; const float* bd;
  const float* bf;
  const float* Wv0t;       // rows 256..258 of Wv0 (3 x 128)
  const float* bv0;
  const float* Wv1;        // 128 x 3
  const float* bv1;
  float* rgb; float* den; float* lat;
};

__device__ __forceinline__ v16h load_a16(const _Float16* p) {
  h8 lo = *(const h8*)p;          // lane-half K pattern, first 8
  h8 hi = *(const h8*)(p + 16);   // +16 block
  return __builtin_shufflevector(lo, hi, 0,1,2,3,4,5,6,7,8,9,10,11,12,13,14,15);
}

__device__ __forceinline__ v8f wmma16(v16h a, v16h b, v8f c) {
  return __builtin_amdgcn_wmma_f32_16x16x32_f16(false, a, false, b, (short)0, c,
                                                false, false);
}

// 4 M-tiles x 2 N-tiles: each B fragment feeds 4 WMMAs, each A fragment 2.
// c[2*mt + j] accumulates (M-tile mt) x (N-tile ntj).
__device__ __forceinline__ void gemm_4x2(const _Float16* __restrict__ xin, int sx,
                                         int ktiles, const _Float16* __restrict__ wB,
                                         int NT, int nt0, int nt1, int kt0, int lane,
                                         v8f c[8]) {
  const _Float16* arow = xin + (lane & 15) * sx + (lane >> 4) * 8;
#pragma unroll
  for (int kt = 0; kt < ktiles; ++kt) {
    v16h a[4];
#pragma unroll
    for (int mt = 0; mt < 4; ++mt) a[mt] = load_a16(arow + mt * 16 * sx + kt * 32);
    const _Float16* wb = wB + (size_t)(kt0 + kt) * NT * 512;   // 512 = 32*16
    v16h b0 = *(const v16h*)(wb + ((size_t)nt0 * 32 + (size_t)lane) * 16);
    v16h b1 = *(const v16h*)(wb + ((size_t)nt1 * 32 + (size_t)lane) * 16);
#pragma unroll
    for (int mt = 0; mt < 4; ++mt) {
      c[2 * mt + 0] = wmma16(a[mt], b0, c[2 * mt + 0]);
      c[2 * mt + 1] = wmma16(a[mt], b1, c[2 * mt + 1]);
    }
  }
}

// 4 M-tiles x 1 N-tile (for the 128-wide Wv0 stage)
__device__ __forceinline__ void gemm_4x1(const _Float16* __restrict__ xin, int sx,
                                         int ktiles, const _Float16* __restrict__ wB,
                                         int NT, int nt, int lane, v8f c[4]) {
#pragma unroll
  for (int kt = 0; kt < ktiles; ++kt) {
    const _Float16* arow = xin + (lane & 15) * sx + (lane >> 4) * 8 + kt * 32;
    v16h b = *(const v16h*)(wB + ((size_t)(kt * NT + nt) * 32 + (size_t)lane) * 16);
#pragma unroll
    for (int mt = 0; mt < 4; ++mt)
      c[mt] = wmma16(load_a16(arow + mt * 16 * sx), b, c[mt]);
  }
}

// C/D layout: m = mt*16 + r + 8*(lane>>4), n = column (precomputed by caller)
__device__ __forceinline__ void store_tile(v8f c, _Float16* xout, int sx, int mt,
                                           int n, int half, float bn, bool relu) {
#pragma unroll
  for (int r = 0; r < 8; ++r) {
    int m = mt * 16 + r + half * 8;
    float v = c[r] + bn;
    if (relu) v = fmaxf(v, 0.f);
    xout[m * sx + n] = (_Float16)v;
  }
}

__global__ void __launch_bounds__(256) mlp_kernel(MlpArgs P) {
  __shared__ _Float16 sFeat[MTILE * SF];
  __shared__ _Float16 sX0[MTILE * SX];
  __shared__ _Float16 sX1[MTILE * SX];

  const int tid  = threadIdx.x;
  const int lane = tid & 31;
  const int wid  = tid >> 5;                // 0..7
  const int half = lane >> 4;
  const size_t p0 = (size_t)blockIdx.x * MTILE;

  // stage feats tile (64 x 128 f16) into LDS: 1024 chunks of 8 f16
#pragma unroll
  for (int e = tid; e < MTILE * 16; e += 256) {
    int m  = e >> 4;
    int cc = (e & 15) * 8;
    *(h8*)&sFeat[m * SF + cc] = *(const h8*)&P.feats[(p0 + (size_t)m) * FEAT + cc];
  }
  __syncthreads();

  const int nt0 = wid, nt1 = wid + 8;
  const int n0 = nt0 * 16 + (lane & 15);
  const int n1 = nt1 * 16 + (lane & 15);

  // layer 0: feats(128) -> sX0(256), relu
  {
    v8f c[8] = {};
    gemm_4x2(sFeat, SF, 4, P.w + kWOff[0], 16, nt0, nt1, 0, lane, c);
    float bn0 = P.b[0][n0], bn1 = P.b[0][n1];
#pragma unroll
    for (int mt = 0; mt < 4; ++mt) {
      store_tile(c[2 * mt + 0], sX0, SX, mt, n0, half, bn0, true);
      store_tile(c[2 * mt + 1], sX0, SX, mt, n1, half, bn1, true);
    }
  }
  __syncthreads();

  // layers 1..7 (layer 4 concatenates feats: K tiles 8..11 of W4)
  _Float16* bufs[2] = { sX0, sX1 };
#pragma unroll
  for (int L = 1; L < 8; ++L) {
    const _Float16* xin  = bufs[(L + 1) & 1];
    _Float16*       xout = bufs[L & 1];
    const _Float16* wB   = P.w + kWOff[L];
    v8f c[8] = {};
    gemm_4x2(xin, SX, 8, wB, 16, nt0, nt1, 0, lane, c);
    if (L == 4)
      gemm_4x2(sFeat, SF, 4, wB, 16, nt0, nt1, 8, lane, c);
    float bn0 = P.b[L][n0], bn1 = P.b[L][n1];
#pragma unroll
    for (int mt = 0; mt < 4; ++mt) {
      store_tile(c[2 * mt + 0], xout, SX, mt, n0, half, bn0, true);
      store_tile(c[2 * mt + 1], xout, SX, mt, n1, half, bn1, true);
    }
    __syncthreads();
  }
  // final x resides in sX1

  // latent = x @ Wf + bf (no relu) -> sX0
  {
    v8f c[8] = {};
    gemm_4x2(sX1, SX, 8, P.w + kWOff[8], 16, nt0, nt1, 0, lane, c);
    float bn0 = P.bf[n0], bn1 = P.bf[n1];
#pragma unroll
    for (int mt = 0; mt < 4; ++mt) {
      store_tile(c[2 * mt + 0], sX0, SX, mt, n0, half, bn0, false);
      store_tile(c[2 * mt + 1], sX0, SX, mt, n1, half, bn1, false);
    }
  }

  // density head (threads 0..63), reads sX1 before it gets recycled
  if (tid < MTILE) {
    float s = P.bd[0];
    for (int k = 0; k < HID; ++k) s += (float)sX1[tid * SX + k] * P.Wd[k];
    P.den[p0 + (size_t)tid] = s;
  }
  __syncthreads();

  // latent -> global f32 (coalesced)
  for (int e = tid; e < MTILE * HID; e += 256) {
    int m = e >> 8, n = e & 255;
    P.lat[(p0 + (size_t)m) * HID + n] = (float)sX0[m * SX + n];
  }

  // h = relu(latent @ Wv0a + bv0 + vd @ Wv0tail) -> sX1 (64 x 128)
  {
    int nt = wid;                            // NT = 8, one N-tile per wave
    int n  = nt * 16 + (lane & 15);
    float w0 = P.Wv0t[0 * 128 + n];
    float w1 = P.Wv0t[1 * 128 + n];
    float w2 = P.Wv0t[2 * 128 + n];
    float bn = P.bv0[n];
    v8f c[4];
#pragma unroll
    for (int mt = 0; mt < 4; ++mt) {         // fold view-dir rows into C init
#pragma unroll
      for (int r = 0; r < 8; ++r) {
        int m = mt * 16 + r + half * 8;
        const float* vda = P.vd + (p0 + (size_t)m) * 3;
        c[mt][r] = bn + vda[0] * w0 + vda[1] * w1 + vda[2] * w2;
      }
    }
    gemm_4x1(sX0, SX, 8, P.w + kWOff[9], 8, nt, lane, c);
#pragma unroll
    for (int mt = 0; mt < 4; ++mt)
      store_tile(c[mt], sX1, SX, mt, n, half, 0.f, true);
  }
  __syncthreads();

  // rgb = sigmoid(h @ Wv1 + bv1), threads 0..191
  if (tid < MTILE * 3) {
    int m = tid / 3, ch = tid % 3;
    float s = P.bv1[ch];
    for (int k = 0; k < 128; ++k) s += (float)sX1[m * SX + k] * P.Wv1[k * 3 + ch];
    P.rgb[(p0 + (size_t)m) * 3 + ch] = 1.f / (1.f + expf(-s));
  }
}

// ---------------------------------------------------------------------------
extern "C" void kernel_launch(void* const* d_in, const int* in_sizes, int n_in,
                              void* d_out, int out_size, void* d_ws, size_t ws_size,
                              hipStream_t stream) {
  (void)in_sizes; (void)n_in; (void)out_size; (void)ws_size;

  const float* coords = (const float*)d_in[0];
  const float* vd     = (const float*)d_in[1];
  const float* g0 = (const float*)d_in[2];
  const float* g1 = (const float*)d_in[3];
  const float* g2 = (const float*)d_in[4];
  const float* g3 = (const float*)d_in[5];
  const float* W[8]; const float* b[8];
  for (int i = 0; i < 8; ++i) { W[i] = (const float*)d_in[6 + i]; b[i] = (const float*)d_in[14 + i]; }
  const float* Wd  = (const float*)d_in[22];
  const float* bd  = (const float*)d_in[23];
  const float* Wf  = (const float*)d_in[24];
  const float* bf  = (const float*)d_in[25];
  const float* Wv0 = (const float*)d_in[26];
  const float* bv0 = (const float*)d_in[27];
  const float* Wv1 = (const float*)d_in[28];
  const float* bv1 = (const float*)d_in[29];

  _Float16* feats = (_Float16*)d_ws;                    // N x 128 f16 (64 MB)
  _Float16* wbuf  = feats + (size_t)NPTS * FEAT;        // swizzled weights (~1.2 MB)

  auto sw = [&](const float* src, int K, int Nout, size_t off) {
    int total = K * Nout;
    swizzle_kernel<<<(total + 255) / 256, 256, 0, stream>>>(src, wbuf + off, K, Nout);
  };
  sw(W[0], 128, 256, kWOff[0]);
  for (int i = 1; i < 8; ++i) sw(W[i], (i == 4) ? 384 : 256, 256, kWOff[i]);
  sw(Wf, 256, 256, kWOff[8]);
  sw(Wv0, 256, 128, kWOff[9]);   // only rows 0..255 are read

  sample_kernel<<<NPTS / 256, 256, 0, stream>>>(coords, g0, g1, g2, g3, feats);

  float* out = (float*)d_out;
  MlpArgs P;
  P.feats = feats; P.w = wbuf; P.vd = vd;
  for (int i = 0; i < 8; ++i) P.b[i] = b[i];
  P.Wd = Wd; P.bd = bd; P.bf = bf;
  P.Wv0t = Wv0 + 256 * 128;
  P.bv0 = bv0; P.Wv1 = Wv1; P.bv1 = bv1;
  P.rgb = out;
  P.den = out + (size_t)NPTS * 3;
  P.lat = out + (size_t)NPTS * 4;

  mlp_kernel<<<NPTS / MTILE, 256, 0, stream>>>(P);
}
